// ParaNNTranSegmentor_55001351193246
// MI455X (gfx1250) — matrix operand order repaired and checked by reading, not compiled, not run on hardware
//
#include <hip/hip_runtime.h>
#include <hip/hip_bf16.h>
#include <stdint.h>

// Sizes (fixed by the reference)
#define B_  256
#define S_  256
#define E_  128
#define HC_ 128
#define HW_ 256

typedef _Float16 half_t;
typedef __attribute__((ext_vector_type(8)))  _Float16 v8h;
typedef __attribute__((ext_vector_type(16))) _Float16 v16h;
typedef __attribute__((ext_vector_type(8)))  float    v8f;

__device__ __forceinline__ float sigf(float x) { return 1.0f / (1.0f + __expf(-x)); }

// ---------------------------------------------------------------------------
// WMMA fragment loaders (CDNA5 ISA §7.12.2, wave32)
// A: 16x32 f16, row-major source. lane l: M = l&15; code's K order, same for B: lanes 0-15 hold K=0..15,
//    lanes 16-31 hold K=16..31  -> two contiguous b128 loads.
// ---------------------------------------------------------------------------
__device__ __forceinline__ v16h load_a_frag(const half_t* __restrict__ A, int lda,
                                            int m0, int k0, int lane) {
  int m  = m0 + (lane & 15);
  int kb = k0 + ((lane >> 4) << 4);
  const half_t* p = A + (size_t)m * lda + kb;
  v8h lo = *(const v8h*)(p);
  v8h hi = *(const v8h*)(p + 8);
  v16h r;
#pragma unroll
  for (int i = 0; i < 8; ++i) { r[i] = lo[i]; r[8 + i] = hi[i]; }
  return r;
}

// B: 32x16 f16 (K x N). Source W is [N, K] row-major (we need B[k][n] = W[n][k]).
// lane l: N = l&15; lanes 0-15 hold K=0..15, lanes 16-31 hold K=16..31
//  -> 16 contiguous halfs along K of one W row: two b128 loads.
__device__ __forceinline__ v16h load_b_frag(const half_t* __restrict__ W, int ldw,
                                            int n0, int k0, int lane) {
  int n  = n0 + (lane & 15);
  int kb = k0 + ((lane >> 4) << 4);
  const half_t* p = W + (size_t)n * ldw + kb;
  v8h lo = *(const v8h*)(p);
  v8h hi = *(const v8h*)(p + 8);
  v16h r;
#pragma unroll
  for (int i = 0; i < 8; ++i) { r[i] = lo[i]; r[8 + i] = hi[i]; }
  return r;
}

// ---------------------------------------------------------------------------
// Gate GEMM:  G[M,N] f32 = A1[M,K1]*W[:,0:K1]^T + A2[M,K2]*W[:,K1:K1+K2]^T
// W is f16 [N, K1+K2] row-major. Each wave computes a 16 x (16*NT) strip:
// the A fragment is loaded once per 32-deep k-block and reused by NT WMMAs.
// K1/K2 are compile-time so the k-loops fully unroll and loads pipeline.
// Grids are exact-fit -> EXEC all-1s for every WMMA.
// ---------------------------------------------------------------------------
template <int K1, int K2, int NT>
__global__ void __launch_bounds__(256)
gates_gemm_t(const half_t* __restrict__ A1, int lda1,
             const half_t* __restrict__ A2, int lda2,
             const half_t* __restrict__ W,  int ldw,
             float* __restrict__ G, int N, int M) {
  int lane = threadIdx.x & 31;
  int wave = threadIdx.x >> 5;
  int gw   = blockIdx.x * 8 + wave;
  int mt   = M >> 4;
  int m0   = (gw % mt) << 4;
  int n0   = (gw / mt) * (16 * NT);

  v8f acc[NT];
#pragma unroll
  for (int i = 0; i < NT; ++i) acc[i] = {};

#pragma unroll
  for (int k = 0; k < K1; k += 32) {
    v16h a = load_a_frag(A1, lda1, m0, k, lane);
    if (k + 32 < K1)   // pull next A k-block toward the WGP (global_prefetch_b8)
      __builtin_prefetch(A1 + (size_t)(m0 + (lane & 15)) * lda1 + k + 32, 0, 0);
#pragma unroll
    for (int i = 0; i < NT; ++i) {
      v16h b = load_b_frag(W, ldw, n0 + 16 * i, k, lane);
      acc[i] = __builtin_amdgcn_wmma_f32_16x16x32_f16(false, a, false, b,
                                                      (short)0, acc[i],
                                                      false, false);
    }
  }
#pragma unroll
  for (int k = 0; k < K2; k += 32) {
    v16h a = load_a_frag(A2, lda2, m0, k, lane);
    if (k + 32 < K2)
      __builtin_prefetch(A2 + (size_t)(m0 + (lane & 15)) * lda2 + k + 32, 0, 0);
#pragma unroll
    for (int i = 0; i < NT; ++i) {
      v16h b = load_b_frag(W, ldw, n0 + 16 * i, K1 + k, lane);
      acc[i] = __builtin_amdgcn_wmma_f32_16x16x32_f16(false, a, false, b,
                                                      (short)0, acc[i],
                                                      false, false);
    }
  }

  // C/D layout: VGPR r, lanes 0-15: (M=r, N=lane); lanes 16-31: (M=8+r, N=lane-16)
  int nl = lane & 15;
  int mb = m0 + ((lane >> 4) << 3);
#pragma unroll
  for (int i = 0; i < NT; ++i) {
#pragma unroll
    for (int r = 0; r < 8; ++r)
      G[(size_t)(mb + r) * N + (n0 + 16 * i + nl)] = acc[i][r];
  }
}

// ---------------------------------------------------------------------------
// Prep: embedding lookup (f32 -> f16), x layout [S, B, E]
// ---------------------------------------------------------------------------
__global__ void __launch_bounds__(256)
embed_kernel(const int* __restrict__ insts, const float* __restrict__ emb,
             half_t* __restrict__ x) {
  int idx = blockIdx.x * 256 + threadIdx.x;    // S*B*E = 8,388,608
  int e = idx & (E_ - 1);
  int b = (idx >> 7) & (B_ - 1);
  int s = idx >> 15;
  int tok = insts[b * S_ + s];                 // insts is [B, S]
  x[idx] = (half_t)emb[(size_t)tok * E_ + e];
}

// Pack all LSTM weights to f16, concatenated [N, Kih + Khh] row-major.
__global__ void __launch_bounds__(256)
pack_weights(const float* __restrict__ WihF, const float* __restrict__ WhhF,
             const float* __restrict__ WihB, const float* __restrict__ WhhB,
             const float* __restrict__ sWih, const float* __restrict__ sWhh,
             const float* __restrict__ wWih, const float* __restrict__ wWhh,
             half_t* __restrict__ Wcf, half_t* __restrict__ Wcb,
             half_t* __restrict__ Wsub, half_t* __restrict__ Wword) {
  int idx = blockIdx.x * 256 + threadIdx.x;    // 1,310,720 total
  if (idx < 262144) {                          // char fwd / bwd: [512, 256]
    int which = idx >> 17;
    int r = idx & 131071;
    int n = r >> 8, k = r & 255;
    const float* Wih = which ? WihB : WihF;
    const float* Whh = which ? WhhB : WhhF;
    float v = (k < 128) ? Wih[n * 128 + k] : Whh[n * 128 + (k - 128)];
    (which ? Wcb : Wcf)[r] = (half_t)v;
  } else {                                     // sub / word: [1024, 512]
    int r2 = idx - 262144;
    int which = (r2 >= 524288) ? 1 : 0;
    int r = which ? (r2 - 524288) : r2;
    int n = r >> 9, k = r & 511;
    const float* Wih = which ? wWih : sWih;
    const float* Whh = which ? wWhh : sWhh;
    float v = (k < 256) ? Wih[n * 256 + k] : Whh[n * 256 + (k - 256)];
    (which ? Wword : Wsub)[r] = (half_t)v;
  }
}

// ---------------------------------------------------------------------------
// Char BiLSTM cell elementwise: both directions. G is [B,512] per dir.
// Writes updated (h,c) state and the chars feature buffer [S, B, 2*HC] f16.
// ---------------------------------------------------------------------------
__global__ void __launch_bounds__(256)
char_cell_elt(const float* __restrict__ Gf, const float* __restrict__ Gb,
              const float* __restrict__ biasF, const float* __restrict__ biasB,
              float* __restrict__ c, half_t* __restrict__ h,
              half_t* __restrict__ chars, int t) {
  int idx = blockIdx.x * 256 + threadIdx.x;    // 2 * B * HC = 65536
  int j = idx & (HC_ - 1);
  int b = (idx >> 7) & (B_ - 1);
  int d = idx >> 15;
  const float* G    = d ? Gb : Gf;
  const float* bias = d ? biasB : biasF;
  const float* Grow = G + (size_t)b * 512;
  float gi = Grow[j]        + bias[j];
  float gf = Grow[128 + j]  + bias[128 + j];
  float gg = Grow[256 + j]  + bias[256 + j];
  float go = Grow[384 + j]  + bias[384 + j];
  size_t sidx = (size_t)d * (B_ * HC_) + (size_t)b * HC_ + j;
  float cc = sigf(gf) * c[sidx] + sigf(gi) * tanhf(gg);
  float hh = sigf(go) * tanhf(cc);
  c[sidx] = cc;
  h[sidx] = (half_t)hh;
  int s = d ? (S_ - 1 - t) : t;                // bwd h at scan step t is chars[S-1-t]
  chars[(size_t)s * (B_ * 2 * HC_) + (size_t)b * (2 * HC_) + d * HC_ + j] = (half_t)hh;
}

// ---------------------------------------------------------------------------
// Subword cell elementwise. Stack collapses: smap[g]∈{1,2} -> next read is
// (h1,c1) if g==0 else zeros.
// ---------------------------------------------------------------------------
__global__ void __launch_bounds__(256)
sub_cell_elt(const float* __restrict__ G, const float* __restrict__ bias,
             const int* __restrict__ golds, int t,
             half_t* __restrict__ h_s, float* __restrict__ c_s,
             half_t* __restrict__ h1) {
  int idx = blockIdx.x * 256 + threadIdx.x;    // B * HW = 65536
  int j = idx & (HW_ - 1);
  int b = idx >> 8;
  const float* Grow = G + (size_t)b * 1024;
  float gi = Grow[j]        + bias[j];
  float gf = Grow[256 + j]  + bias[256 + j];
  float gg = Grow[512 + j]  + bias[512 + j];
  float go = Grow[768 + j]  + bias[768 + j];
  float c1 = sigf(gf) * c_s[idx] + sigf(gi) * tanhf(gg);
  float hh = sigf(go) * tanhf(c1);
  h1[idx] = (half_t)hh;
  int g = golds[b * S_ + t];                   // golds is [B, S]
  if (g == 0) { h_s[idx] = (half_t)hh; c_s[idx] = c1; }
  else        { h_s[idx] = (half_t)0;  c_s[idx] = 0.0f; }
}

// ---------------------------------------------------------------------------
// Word cell + classifier (fused, LDS reduce). wmap[g]∈{0,1}: keep old read
// value if g==0, else (h2,c2). out is [B, S, 2] f32.
// ---------------------------------------------------------------------------
__global__ void __launch_bounds__(256)
word_cell_cls(const float* __restrict__ G, const float* __restrict__ bias,
              const int* __restrict__ golds, int t,
              half_t* __restrict__ h_w, float* __restrict__ c_w,
              const half_t* __restrict__ chars,
              const float* __restrict__ clsW, const float* __restrict__ clsb,
              float* __restrict__ out) {
  __shared__ float r0[256];
  __shared__ float r1[256];
  int b = blockIdx.x;
  int j = threadIdx.x;                         // j < HW
  const float* Grow = G + (size_t)b * 1024;
  float gi = Grow[j]        + bias[j];
  float gf = Grow[256 + j]  + bias[256 + j];
  float gg = Grow[512 + j]  + bias[512 + j];
  float go = Grow[768 + j]  + bias[768 + j];
  int sidx = b * HW_ + j;
  float c2 = sigf(gf) * c_w[sidx] + sigf(gi) * tanhf(gg);
  float h2 = sigf(go) * tanhf(c2);
  int g = golds[b * S_ + t];
  if (g != 0) { h_w[sidx] = (half_t)h2; c_w[sidx] = c2; }
  // classifier: concat(h2[256], chars_t[256]) @ clsW[2,512]^T + clsb
  float xc = (float)chars[(size_t)t * (B_ * 2 * HC_) + (size_t)b * (2 * HC_) + j];
  r0[j] = h2 * clsW[j]       + xc * clsW[256 + j];
  r1[j] = h2 * clsW[512 + j] + xc * clsW[768 + j];
  for (int sft = 128; sft > 0; sft >>= 1) {
    __syncthreads();
    if (j < sft) { r0[j] += r0[j + sft]; r1[j] += r1[j + sft]; }
  }
  __syncthreads();
  if (j == 0) {
    out[(size_t)b * (S_ * 2) + t * 2 + 0] = r0[0] + clsb[0];
    out[(size_t)b * (S_ * 2) + t * 2 + 1] = r1[0] + clsb[1];
  }
}

// ---------------------------------------------------------------------------
// Host launcher
// ---------------------------------------------------------------------------
extern "C" void kernel_launch(void* const* d_in, const int* in_sizes, int n_in,
                              void* d_out, int out_size, void* d_ws, size_t ws_size,
                              hipStream_t stream) {
  (void)in_sizes; (void)n_in; (void)out_size; (void)ws_size;
  const int*   insts = (const int*)d_in[0];
  const int*   golds = (const int*)d_in[1];
  const float* emb   = (const float*)d_in[2];
  const float* WihF  = (const float*)d_in[3];
  const float* WhhF  = (const float*)d_in[4];
  const float* bF    = (const float*)d_in[5];
  const float* WihB  = (const float*)d_in[6];
  const float* WhhB  = (const float*)d_in[7];
  const float* bB    = (const float*)d_in[8];
  const float* sWih  = (const float*)d_in[9];
  const float* sWhh  = (const float*)d_in[10];
  const float* sB    = (const float*)d_in[11];
  const float* wWih  = (const float*)d_in[12];
  const float* wWhh  = (const float*)d_in[13];
  const float* wB    = (const float*)d_in[14];
  const float* clsW  = (const float*)d_in[15];
  const float* clsb  = (const float*)d_in[16];
  float* out = (float*)d_out;

  // Workspace carve-up (256B aligned)
  uint8_t* ws = (uint8_t*)d_ws;
  size_t off = 0;
  auto alloc = [&](size_t bytes) -> void* {
    void* p = ws + off;
    off = (off + bytes + 255) & ~(size_t)255;
    return p;
  };
  half_t* x16   = (half_t*)alloc((size_t)S_ * B_ * E_ * 2);          // 16 MB
  half_t* chars = (half_t*)alloc((size_t)S_ * B_ * 2 * HC_ * 2);     // 32 MB
  half_t* Wcf   = (half_t*)alloc((size_t)512 * 256 * 2);
  half_t* Wcb   = (half_t*)alloc((size_t)512 * 256 * 2);
  half_t* Wsub  = (half_t*)alloc((size_t)1024 * 512 * 2);
  half_t* Wword = (half_t*)alloc((size_t)1024 * 512 * 2);
  float*  Gcf   = (float*)alloc((size_t)B_ * 512 * 4);
  float*  Gcb   = (float*)alloc((size_t)B_ * 512 * 4);
  float*  Gsub  = (float*)alloc((size_t)B_ * 1024 * 4);
  float*  Gword = (float*)alloc((size_t)B_ * 1024 * 4);
  // --- zero-initialized state region (contiguous, memset below) ---
  uint8_t* state0 = ws + off;
  half_t* hchar = (half_t*)alloc((size_t)2 * B_ * HC_ * 2);
  float*  cchar = (float*)alloc((size_t)2 * B_ * HC_ * 4);
  half_t* h_s   = (half_t*)alloc((size_t)B_ * HW_ * 2);
  float*  c_s   = (float*)alloc((size_t)B_ * HW_ * 4);
  half_t* h1    = (half_t*)alloc((size_t)B_ * HW_ * 2);
  half_t* h_w   = (half_t*)alloc((size_t)B_ * HW_ * 2);
  float*  c_w   = (float*)alloc((size_t)B_ * HW_ * 4);
  half_t* zpage = (half_t*)alloc((size_t)B_ * 2 * HC_ * 2);  // stays all-zero
  size_t stateBytes = (size_t)((ws + off) - state0);

  // Prep: embedding, weight packing, zero recurrent state (capture-safe).
  embed_kernel<<<(S_ * B_ * E_) / 256, 256, 0, stream>>>(insts, emb, x16);
  pack_weights<<<1310720 / 256, 256, 0, stream>>>(WihF, WhhF, WihB, WhhB,
                                                  sWih, sWhh, wWih, wWhh,
                                                  Wcf, Wcb, Wsub, Wword);
  hipMemsetAsync(state0, 0, stateBytes, stream);

  half_t* hcf = hchar;
  half_t* hcb = hchar + (size_t)B_ * HC_;

  // --- Char BiLSTM: 256 sequential steps, fwd & bwd GEMMs independent ---
  // [256,512]: 16 m-tiles x 8 n-strips (NT=4) = 128 waves = 16 blocks
  for (int t = 0; t < S_; ++t) {
    const half_t* xf = x16 + (size_t)t * B_ * E_;
    const half_t* xb = x16 + (size_t)(S_ - 1 - t) * B_ * E_;
    gates_gemm_t<128, 128, 4><<<16, 256, 0, stream>>>(xf, E_, hcf, HC_,
                                                      Wcf, 256, Gcf, 512, B_);
    gates_gemm_t<128, 128, 4><<<16, 256, 0, stream>>>(xb, E_, hcb, HC_,
                                                      Wcb, 256, Gcb, 512, B_);
    char_cell_elt<<<256, 256, 0, stream>>>(Gcf, Gcb, bF, bB, cchar, hchar,
                                           chars, t);
  }

  // --- Transition scan: subword cell -> word cell -> classifier per step ---
  // [256,1024]: 16 m-tiles x 16 n-strips (NT=4) = 256 waves = 32 blocks
  for (int t = 0; t < S_; ++t) {
    const half_t* subin = (t == 0) ? zpage
                                   : chars + (size_t)(t - 1) * B_ * 2 * HC_;
    gates_gemm_t<256, 256, 4><<<32, 256, 0, stream>>>(subin, 2 * HC_, h_s, HW_,
                                                      Wsub, 512, Gsub, 1024, B_);
    sub_cell_elt<<<256, 256, 0, stream>>>(Gsub, sB, golds, t, h_s, c_s, h1);
    gates_gemm_t<256, 256, 4><<<32, 256, 0, stream>>>(h1, HW_, h_w, HW_,
                                                      Wword, 512, Gword, 1024, B_);
    word_cell_cls<<<B_, 256, 0, stream>>>(Gword, wB, golds, t, h_w, c_w,
                                          chars, clsW, clsb, out);
  }
}
